// MultiTaskProjector_42296837931177
// MI455X (gfx1250) — matrix operand order repaired and checked by reading, not compiled, not run on hardware
//
#include <hip/hip_runtime.h>
#include <hip/hip_bf16.h>

typedef __attribute__((ext_vector_type(16))) __bf16 v16bf;
typedef __attribute__((ext_vector_type(8)))  float  v8f;

union FragBF { v16bf v; uint4 q[2]; };
union AccU   { v8f  v; float f[8]; };

__device__ __forceinline__ unsigned short f2bf(float f) {
  unsigned u = __float_as_uint(f);
  u += 0x7FFFu + ((u >> 16) & 1u);          // round-to-nearest-even
  return (unsigned short)(u >> 16);
}
__device__ __forceinline__ float bf2f(unsigned short h) {
  return __uint_as_float(((unsigned)h) << 16);
}

// ---------------------------------------------------------------------------
// bf16 weight pack: Wt[co][tap*Cin + cin] = w[co][cin][ky][kx]
// ---------------------------------------------------------------------------
__global__ __launch_bounds__(256) void pack_w_kernel(
    const float* __restrict__ w, unsigned short* __restrict__ wt,
    int Cout, int Cin) {
  int idx = blockIdx.x * 256 + threadIdx.x;
  int n = Cout * Cin * 9;
  if (idx >= n) return;
  int cin = idx % Cin;
  int r   = idx / Cin;
  int tap = r % 9;
  int co  = r / 9;
  wt[idx] = f2bf(w[((size_t)(co * Cin + cin)) * 9 + tap]);
}

// ---------------------------------------------------------------------------
// text branch: wb[b][o] = word[b] . txt_w[o] + txt_b[o]   (8 x 2305 x 1024)
// ---------------------------------------------------------------------------
__global__ __launch_bounds__(256) void txt_gemm_kernel(
    const float* __restrict__ word, const float* __restrict__ tw,
    const float* __restrict__ tb, float* __restrict__ wb) {
  int idx = blockIdx.x * 256 + threadIdx.x;
  if (idx >= 8 * 2305) return;
  int o = idx % 2305, b = idx / 2305;
  const float* wr = word + (size_t)b * 1024;
  const float* tr = tw + (size_t)o * 1024;
  float s = 0.f;
  for (int d = 0; d < 1024; ++d) s = fmaf(wr[d], tr[d], s);
  wb[idx] = s + tb[o];
}

// ---------------------------------------------------------------------------
// W_eff[b][tap][t][c2] = sum_c dyn_w[b][c][tap] * w3[t*256+c][c2]
// (fuses the 1x1 conv3 into the dynamic per-batch 3x3 conv)
// ---------------------------------------------------------------------------
__global__ __launch_bounds__(256) void weff_kernel(
    const float* __restrict__ wb, const float* __restrict__ w3,
    float* __restrict__ weff) {
  int idx = blockIdx.x * 256 + threadIdx.x;
  if (idx >= 8 * 9 * 5 * 256) return;
  int c2  = idx & 255;
  int t5  = (idx >> 8) % 5;
  int tap = (idx / 1280) % 9;
  int b   = idx / 11520;
  const float* wrow = wb + (size_t)b * 2305;
  float s = 0.f;
  for (int c = 0; c < 256; ++c)
    s = fmaf(wrow[c * 9 + tap], w3[((size_t)(t5 * 256 + c)) * 256 + c2], s);
  weff[idx] = s;
}

// tapbias[t][b][tap] = sum_c dyn_w[b][c][tap]*b3[t*256+c] ;  dynb[b]
__global__ void tapbias_kernel(const float* __restrict__ wb,
                               const float* __restrict__ b3,
                               float* __restrict__ tapb,
                               float* __restrict__ dynb) {
  int idx = threadIdx.x;  // single block of 512
  if (idx < 360) {
    int tap = idx % 9;
    int b   = (idx / 9) % 8;
    int t5  = idx / 72;
    const float* wrow = wb + (size_t)b * 2305;
    float s = 0.f;
    for (int c = 0; c < 256; ++c)
      s = fmaf(wrow[c * 9 + tap], b3[t5 * 256 + c], s);
    tapb[(t5 * 8 + b) * 9 + tap] = s;
  } else if (idx < 368) {
    int b = idx - 360;
    dynb[b] = wb[(size_t)b * 2305 + 2304];
  }
}

// ---------------------------------------------------------------------------
// 2x bilinear upsample (half-pixel centers, edge clamp).
// Thread order is (b, oy, ox, c) == NHWC-linear output -> coalesced writes.
// ---------------------------------------------------------------------------
__device__ __forceinline__ void ups_coords(int o, int In, int& i0, int& i1,
                                           float& w0) {
  int j = o >> 1;
  if ((o & 1) == 0) { i0 = j > 0 ? j - 1 : 0; i1 = j; w0 = 0.25f; }
  else              { i0 = j; i1 = (j + 1 < In) ? j + 1 : In - 1; w0 = 0.75f; }
}

// x: NCHW f32 -> NHWC bf16
__global__ __launch_bounds__(256) void ups_nchw_f32_kernel(
    const float* __restrict__ in, unsigned short* __restrict__ out,
    int B, int C, int Hin, int Win) {
  int Hout = Hin * 2, Wout = Win * 2;
  size_t idx = (size_t)blockIdx.x * 256 + threadIdx.x;
  size_t total = (size_t)B * Hout * Wout * C;
  if (idx >= total) return;
  int c  = (int)(idx % C);
  size_t r = idx / C;
  int ox = (int)(r % Wout);
  r /= Wout;
  int oy = (int)(r % Hout);
  int b  = (int)(r / Hout);
  int y0, y1, x0, x1; float wy0, wx0;
  ups_coords(oy, Hin, y0, y1, wy0);
  ups_coords(ox, Win, x0, x1, wx0);
  float wy1 = 1.f - wy0, wx1 = 1.f - wx0;
  const float* p = in + ((size_t)(b * C + c) * Hin) * Win;
  float v = wy0 * (wx0 * p[y0 * Win + x0] + wx1 * p[y0 * Win + x1])
          + wy1 * (wx0 * p[y1 * Win + x0] + wx1 * p[y1 * Win + x1]);
  out[idx] = f2bf(v);
}

// a1: NHWC bf16 -> NHWC bf16 (fully coalesced both sides)
__global__ __launch_bounds__(256) void ups_nhwc_bf16_kernel(
    const unsigned short* __restrict__ in, unsigned short* __restrict__ out,
    int B, int C, int Hin, int Win) {
  int Hout = Hin * 2, Wout = Win * 2;
  size_t idx = (size_t)blockIdx.x * 256 + threadIdx.x;
  size_t total = (size_t)B * Hout * Wout * C;
  if (idx >= total) return;
  int c  = (int)(idx % C);
  size_t r = idx / C;
  int ox = (int)(r % Wout);
  r /= Wout;
  int oy = (int)(r % Hout);
  int b  = (int)(r / Hout);
  int y0, y1, x0, x1; float wy0, wx0;
  ups_coords(oy, Hin, y0, y1, wy0);
  ups_coords(ox, Win, x0, x1, wx0);
  float wy1 = 1.f - wy0, wx1 = 1.f - wx0;
  const unsigned short* p = in + (size_t)b * Hin * Win * C + c;
  float v = wy0 * (wx0 * bf2f(p[(y0 * Win + x0) * C]) + wx1 * bf2f(p[(y0 * Win + x1) * C]))
          + wy1 * (wx0 * bf2f(p[(y1 * Win + x0) * C]) + wx1 * bf2f(p[(y1 * Win + x1) * C]));
  out[idx] = f2bf(v);
}

// ---------------------------------------------------------------------------
// Implicit-GEMM 3x3 conv (pad=1, Cin=512) + BN + ReLU, bf16 NHWC in/out,
// f32 acc. Block tile: M=64 pixels x N=64 couts; 8 waves = 2 M-supers x
// 4 N-tiles; each wave holds TWO 16x16 accumulators (B fragment reused by
// both -> 4 v_wmma per K-64 step). Double-buffered LDS A tile: one barrier
// per step; fill addressing is fully branchless (clamped coords + cndmask
// zeroing) and all step bookkeeping is incremental (no div/mod in loop).
// ---------------------------------------------------------------------------
__global__ __launch_bounds__(256) void conv3x3_wmma_kernel(
    const unsigned short* __restrict__ in,   // bf16 NHWC (B, H, W, 512)
    const unsigned short* __restrict__ wt,   // bf16 [Cout][9*512] tap-major
    const float* __restrict__ bng, const float* __restrict__ bnb,
    const float* __restrict__ bnm, const float* __restrict__ bnv,
    unsigned short* __restrict__ out,        // bf16 NHWC (B, H, W, Cout)
    int Cout, int H, int W) {
  constexpr int CIN  = 512;
  constexpr int KTOT = CIN * 9;
  constexpr int LDR  = 72;       // LDS row stride (halfwords), padded
  const int HW  = H * W;
  const int bb  = blockIdx.z;
  const int co0 = blockIdx.y * 64;
  const int p0  = blockIdx.x * 64;

  __shared__ __align__(16) unsigned short As[2][64 * LDR];  // 2 x 9216 B

  const int t    = threadIdx.x;
  const int lane = t & 31;
  const int wid  = t >> 5;
  const int wn   = wid & 3;   // cout tile
  const int wm   = wid >> 2;  // pixel super-tile (32 px)
  const int ln16 = lane & 15;
  const int half = lane >> 4;

  // fill role: pixel pl (0..63), channel group kg (16 contiguous channels)
  const int pl = t & 63;
  const int kg = t >> 6;
  const int pf = p0 + pl;
  const int fy = pf / W;
  const int fx = pf - fy * W;

  const unsigned short* inB = in + (size_t)bb * HW * CIN;

  AccU acc0, acc1;
#pragma unroll
  for (int i = 0; i < 8; ++i) { acc0.f[i] = 0.f; acc1.f[i] = 0.f; }

  const int co = co0 + wn * 16 + ln16;
  const unsigned short* wrow = wt + (size_t)co * KTOT;

  // Branchless tile fill: clamped coords make every address valid; invalid
  // taps are zeroed with cndmask on the loaded data.
  auto fillf = [&](int dy, int dx, int cbv, int buf) {
    const int iy = fy + dy, ix = fx + dx;
    const bool v = (pf < HW) & (iy >= 0) & (iy < H) & (ix >= 0) & (ix < W);
    const int iyc = min(max(iy, 0), H - 1);
    const int ixc = min(max(ix, 0), W - 1);
    const int spix = iyc * W + ixc;
    const unsigned short* src = inB + (size_t)spix * CIN + cbv + kg * 16;
    uint4 q0 = *(const uint4*)(src);
    uint4 q1 = *(const uint4*)(src + 8);
    if (!v) {
      q0.x = 0; q0.y = 0; q0.z = 0; q0.w = 0;
      q1.x = 0; q1.y = 0; q1.z = 0; q1.w = 0;
    }
    unsigned short* d = &As[buf][pl * LDR + kg * 16];
    *(uint4*)d = q0;
    *(uint4*)(d + 8) = q1;
  };

  // steps: 9 taps x 8 (512/64) channel blocks = 72
  fillf(-1, -1, 0, 0);  // step 0
  __syncthreads();

  // next-fill coords (step 1): tap 0 (dy=-1,dx=-1), cb=64
  int ndy = -1, ndx = -1, ncb = 64;
  const unsigned short* abase0 = &As[0][(wm * 32 + ln16) * LDR + half * 8];
  const unsigned short* abase1 = &As[1][(wm * 32 + ln16) * LDR + half * 8];
  const unsigned short* bstep = wrow + half * 8;  // advances 64 hw per step

  for (int s = 0; s < 72; ++s) {
    const int cur = s & 1;
    if (s < 71) fillf(ndy, ndx, ncb, cur ^ 1);  // prefetch next tile

    const unsigned short* abase = cur ? abase1 : abase0;
#pragma unroll
    for (int j = 0; j < 2; ++j) {        // two 32-wide k-blocks
      FragBF af0, af1, bf;
      const unsigned short* ap = abase + j * 32;
      af0.q[0] = *(const uint4*)(ap);
      af0.q[1] = *(const uint4*)(ap + 16);
      af1.q[0] = *(const uint4*)(ap + 16 * LDR);
      af1.q[1] = *(const uint4*)(ap + 16 * LDR + 16);
      const unsigned short* bp = bstep + j * 32;
      bf.q[0] = *(const uint4*)(bp);
      bf.q[1] = *(const uint4*)(bp + 16);
      acc0.v = __builtin_amdgcn_wmma_f32_16x16x32_bf16(
          false, af0.v, false, bf.v, (short)0, acc0.v, false, false);
      acc1.v = __builtin_amdgcn_wmma_f32_16x16x32_bf16(
          false, af1.v, false, bf.v, (short)0, acc1.v, false, false);
    }
    __builtin_prefetch(bstep + 64, 0, 3);  // next K-block of weights
    __syncthreads();

    // incremental bookkeeping (uniform scalar)
    bstep += 64;
    ncb += 64;
    if (ncb == CIN) {
      ncb = 0;
      ++ndx;
      if (ndx == 2) { ndx = -1; ++ndy; }
    }
  }

  // BN + ReLU epilogue (per-lane cout fixed), NHWC store
  const float ivs = rsqrtf(bnv[co] + 1e-5f);
  const float s   = bng[co] * ivs;
  const float sh  = bnb[co] - bnm[co] * s;
  const int pb = p0 + wm * 32 + half * 8;
#pragma unroll
  for (int r = 0; r < 8; ++r) {
    const int pA = pb + r;
    if (pA < HW) {
      float v = fmaf(acc0.f[r], s, sh);
      v = v > 0.f ? v : 0.f;
      out[((size_t)bb * HW + pA) * Cout + co] = f2bf(v);
    }
    const int pB = pb + 16 + r;
    if (pB < HW) {
      float v = fmaf(acc1.f[r], s, sh);
      v = v > 0.f ? v : 0.f;
      out[((size_t)bb * HW + pB) * Cout + co] = f2bf(v);
    }
  }
}

// ---------------------------------------------------------------------------
// Fused (conv3 + dynamic per-batch 3x3 conv), NHWC a2: tap-outer loop with
// uint4 (8xbf16) activation loads; W_eff slab for this batch staged in LDS
// as [tap][t5][c2] so every access stream is contiguous.
// ---------------------------------------------------------------------------
__global__ __launch_bounds__(256) void final_dynconv_kernel(
    const unsigned short* __restrict__ a2,  // bf16 NHWC (8,H,W,256)
    const float* __restrict__ weff,         // [b][tap][t5][256]
    const float* __restrict__ tapb,         // [t][b][9]
    const float* __restrict__ dynb,         // [b]
    float* __restrict__ out,                // [t][b][H*W]
    int H, int W) {
  const int HW = H * W;
  const int b = blockIdx.y;
  __shared__ float Wl[9 * 5 * 256];  // 46080 B
  for (int i = threadIdx.x; i < 9 * 5 * 256; i += 256)
    Wl[i] = weff[(size_t)b * (9 * 5 * 256) + i];
  __syncthreads();

  const int p = blockIdx.x * 256 + threadIdx.x;
  if (p >= HW) return;
  const int y = p / W, x = p - y * W;

  float acc[5];
#pragma unroll
  for (int t5 = 0; t5 < 5; ++t5) acc[t5] = dynb[b];

  for (int tap = 0; tap < 9; ++tap) {
    const int dy = tap / 3 - 1, dx = tap % 3 - 1;
    const int yy = y + dy, xx = x + dx;
    if (yy < 0 || yy >= H || xx < 0 || xx >= W) continue;
    const uint4* row =
        (const uint4*)(a2 + ((size_t)b * HW + (yy * W + xx)) * 256);
    const float* wl = &Wl[tap * 5 * 256];
    for (int c8 = 0; c8 < 32; ++c8) {
      const uint4 q = row[c8];
      float f[8];
      f[0] = __uint_as_float(q.x << 16);
      f[1] = __uint_as_float(q.x & 0xFFFF0000u);
      f[2] = __uint_as_float(q.y << 16);
      f[3] = __uint_as_float(q.y & 0xFFFF0000u);
      f[4] = __uint_as_float(q.z << 16);
      f[5] = __uint_as_float(q.z & 0xFFFF0000u);
      f[6] = __uint_as_float(q.w << 16);
      f[7] = __uint_as_float(q.w & 0xFFFF0000u);
#pragma unroll
      for (int t5 = 0; t5 < 5; ++t5) {
        const float* w8 = &wl[t5 * 256 + c8 * 8];
#pragma unroll
        for (int j = 0; j < 8; ++j) acc[t5] = fmaf(w8[j], f[j], acc[t5]);
      }
    }
#pragma unroll
    for (int t5 = 0; t5 < 5; ++t5)
      acc[t5] += tapb[(t5 * 8 + b) * 9 + tap];  // valid taps only (zero-pad of h)
  }

#pragma unroll
  for (int t5 = 0; t5 < 5; ++t5)
    out[((size_t)t5 * 8 + b) * HW + p] = acc[t5];
}

// ---------------------------------------------------------------------------
extern "C" void kernel_launch(void* const* d_in, const int* in_sizes, int n_in,
                              void* d_out, int out_size, void* d_ws, size_t ws_size,
                              hipStream_t stream) {
  (void)in_sizes; (void)n_in; (void)out_size; (void)ws_size;
  const float* x    = (const float*)d_in[0];
  const float* word = (const float*)d_in[1];
  const float* w1   = (const float*)d_in[2];
  const float* bn1g = (const float*)d_in[3];
  const float* bn1b = (const float*)d_in[4];
  const float* bn1m = (const float*)d_in[5];
  const float* bn1v = (const float*)d_in[6];
  const float* w2   = (const float*)d_in[7];
  const float* bn2g = (const float*)d_in[8];
  const float* bn2b = (const float*)d_in[9];
  const float* bn2m = (const float*)d_in[10];
  const float* bn2v = (const float*)d_in[11];
  const float* w3   = (const float*)d_in[12];
  const float* b3   = (const float*)d_in[13];
  const float* txtw = (const float*)d_in[14];
  const float* txtb = (const float*)d_in[15];
  float* out = (float*)d_out;

  char* ws = (char*)d_ws;
  size_t off = 0;
  auto alloc = [&](size_t bytes) {
    char* p = ws + off;
    off = (off + bytes + 255) & ~(size_t)255;
    return p;
  };
  unsigned short* Wt1 = (unsigned short*)alloc(512ull * 4608 * 2);
  unsigned short* Wt2 = (unsigned short*)alloc(256ull * 4608 * 2);
  float* wb   = (float*)alloc(8ull * 2305 * 4);
  float* Weff = (float*)alloc(8ull * 9 * 5 * 256 * 4);
  float* tapb = (float*)alloc(360ull * 4);
  float* dynb = (float*)alloc(8ull * 4);
  unsigned short* u1 = (unsigned short*)alloc(8ull * 52 * 52 * 512 * 2);    // NHWC
  unsigned short* a1 = (unsigned short*)alloc(8ull * 52 * 52 * 512 * 2);    // NHWC
  unsigned short* u2 = (unsigned short*)alloc(8ull * 104 * 104 * 512 * 2);  // NHWC
  unsigned short* a2 = (unsigned short*)alloc(8ull * 104 * 104 * 256 * 2);  // NHWC

  // weight packing (bf16, tap-major K)
  {
    int n1 = 512 * 512 * 9;
    pack_w_kernel<<<(n1 + 255) / 256, 256, 0, stream>>>(w1, Wt1, 512, 512);
    int n2 = 256 * 512 * 9;
    pack_w_kernel<<<(n2 + 255) / 256, 256, 0, stream>>>(w2, Wt2, 256, 512);
  }
  // text branch + dynamic-weight precontraction
  txt_gemm_kernel<<<(8 * 2305 + 255) / 256, 256, 0, stream>>>(word, txtw, txtb, wb);
  weff_kernel<<<(8 * 9 * 5 * 256 + 255) / 256, 256, 0, stream>>>(wb, w3, Weff);
  tapbias_kernel<<<1, 512, 0, stream>>>(wb, b3, tapb, dynb);

  // up2x(x): NCHW f32 -> NHWC bf16 (8,52,52,512)
  {
    size_t tot = 8ull * 52 * 52 * 512;
    ups_nchw_f32_kernel<<<(unsigned)((tot + 255) / 256), 256, 0, stream>>>(
        x, u1, 8, 512, 26, 26);
  }
  // conv1 + BN + ReLU -> a1 NHWC bf16
  {
    dim3 grid((52 * 52 + 63) / 64, 512 / 64, 8);
    conv3x3_wmma_kernel<<<grid, 256, 0, stream>>>(
        u1, Wt1, bn1g, bn1b, bn1m, bn1v, a1, 512, 52, 52);
  }
  // up2x(a1): NHWC bf16 -> NHWC bf16 (8,104,104,512)
  {
    size_t tot = 8ull * 104 * 104 * 512;
    ups_nhwc_bf16_kernel<<<(unsigned)((tot + 255) / 256), 256, 0, stream>>>(
        a1, u2, 8, 512, 52, 52);
  }
  // conv2 + BN + ReLU -> a2 NHWC bf16 (8,104,104,256)
  {
    dim3 grid((104 * 104 + 63) / 64, 256 / 64, 8);
    conv3x3_wmma_kernel<<<grid, 256, 0, stream>>>(
        u2, Wt2, bn2g, bn2b, bn2m, bn2v, a2, 512, 104, 104);
  }
  // fused conv3 + dynamic conv -> out (5,8,1,104,104)
  {
    dim3 grid((104 * 104 + 255) / 256, 8);
    final_dynconv_kernel<<<grid, 256, 0, stream>>>(a2, Weff, tapb, dynb, out, 104, 104);
  }
}
